// SparseHashedNNDistance_68401649156587
// MI455X (gfx1250) — matrix operand order repaired and checked by reading, not compiled, not run on hardware
//
#include <hip/hip_runtime.h>

// ---------------------------------------------------------------------------
// SparseHashedNNDistance for MI455X (gfx1250, wave32, WMMA bf16 16x16x32)
// ---------------------------------------------------------------------------
#define B_     4
#define N_     8000
#define FIN_   256
#define DDIST_ 128
#define DFF_   128
#define NBINS_ 16
#define BSZ_   500
#define K_     8
#define EDGES_ (B_ * N_ * K_)      // 256000
#define ROTC_  100                 // MAX_NUM_BINS/2 (rot row stride)

typedef __attribute__((ext_vector_type(16))) __bf16 v16bf;
typedef __attribute__((ext_vector_type(8)))  __bf16 v8bf;
typedef __attribute__((ext_vector_type(8)))  float  v8f;
typedef __attribute__((ext_vector_type(4)))  float  v4f;

__device__ __forceinline__ float eluf(float x) {
  return x > 0.0f ? x : (expf(x) - 1.0f);
}

__device__ __forceinline__ v16bf cat8(v8bf a, v8bf b) {
  return __builtin_shufflevector(a, b, 0, 1, 2, 3, 4, 5, 6, 7,
                                 8, 9, 10, 11, 12, 13, 14, 15);
}

// D = A(16x32) * B(32x16) + C, bf16 in / f32 accum.
__device__ __forceinline__ v8f wmma_bf16(v16bf a, v16bf b, v8f c) {
  return __builtin_amdgcn_wmma_f32_16x16x32_bf16(false, a, false, b,
                                                 (short)0, c, false, false);
}

// A-fragment (16x32 bf16) from bf16 row-major LDS tile; 2x ds_load_b128.
// lanes 0-15: rows 0-15, K = {k0..k0+7, k0+16..k0+23}
// lanes16-31: rows 0-15, K = {k0+8..k0+15, k0+24..k0+31}
__device__ __forceinline__ v16bf a_frag_lds(const __bf16* base, int ld,
                                            int k0, int lane) {
  const __bf16* p = base + (lane & 15) * ld + k0 + (lane >> 4) * 8;
  const v8bf* q = (const v8bf*)p;          // 16B aligned (k-offsets mult of 8)
  return cat8(q[0], q[2]);                 // p[0..7], p[16..23]
}

// A-fragment converted on the fly from f32 global rows; 4x global_load_b128.
__device__ __forceinline__ v16bf a_frag_f32(const float* base, int ld,
                                            int k0, int lane) {
  const float* p = base + (lane & 15) * ld + k0 + (lane >> 4) * 8;
  const v4f* q = (const v4f*)p;
  v4f x0 = q[0], x1 = q[1], x2 = q[4], x3 = q[5];  // f32 0..7 and 16..23
  v16bf a;
#pragma unroll
  for (int i = 0; i < 4; ++i) {
    a[i]      = (__bf16)x0[i];
    a[4 + i]  = (__bf16)x1[i];
    a[8 + i]  = (__bf16)x2[i];
    a[12 + i] = (__bf16)x3[i];
  }
  return a;
}

// ---- swizzled B-fragment storage: swz[((kt*NT+nt)*32+lane)*16 + i] --------
// Each lane's fragment is 32 contiguous bytes -> 2x ds_load_b128, no packing.
__device__ __forceinline__ v16bf b_frag_swz(const __bf16* swz, int NT, int kt,
                                            int nt, int lane) {
  const v8bf* p = (const v8bf*)(swz + (size_t)((kt * NT + nt) * 32 + lane) * 16);
  return cat8(p[0], p[1]);
}

// Stage f32 row-major weights [K][NT*16] into swizzled bf16 B-fragment order.
__device__ void stage_b_swz(const float* __restrict__ w, __bf16* swz,
                            int K, int NT, int ld) {
  int entries = (K / 32) * NT * 32;        // (#frags) * 32 lanes
  for (int idx = threadIdx.x; idx < entries; idx += blockDim.x) {
    int frag = idx >> 5, lane = idx & 31;
    int kt = frag / NT, nt = frag % NT;
    int krow = kt * 32 + (lane >> 4) * 8;
    int col = nt * 16 + (lane & 15);
    v8bf lo, hi;
#pragma unroll
    for (int i = 0; i < 8; ++i) {
      lo[i] = (__bf16)w[(size_t)(krow + i) * ld + col];
      hi[i] = (__bf16)w[(size_t)(krow + 16 + i) * ld + col];
    }
    v8bf* dst = (v8bf*)(swz + (size_t)idx * 16);
    dst[0] = lo;
    dst[1] = hi;
  }
}

// B-fragment that is the TRANSPOSE of a row-major bf16 tile (for A*A^T):
// element i = T[n0+(lane&15)][k0+(lane>>4)*8+{0..7,16..23}] -- 2x b128.
__device__ __forceinline__ v16bf bT_frag_lds(const __bf16* t, int ld,
                                             int k0, int n0, int lane) {
  const __bf16* p = t + (n0 + (lane & 15)) * ld + k0 + ((lane >> 4) << 3);
  const v8bf* q = (const v8bf*)p;
  return cat8(q[0], q[2]);
}

__device__ __forceinline__ void top8_insert(float (&tv)[8], int (&ti)[8],
                                            float v, int c) {
  if (v <= tv[7]) return;
  tv[7] = v; ti[7] = c;
#pragma unroll
  for (int k = 7; k > 0; --k) {
    if (tv[k] > tv[k - 1]) {
      float tf = tv[k]; tv[k] = tv[k - 1]; tv[k - 1] = tf;
      int   tt = ti[k]; ti[k] = ti[k - 1]; ti[k - 1] = tt;
    }
  }
}

// ---------------------------------------------------------------------------
// K1: point encoder  pe = elu(x @ W1 + b1) @ W2 + b2   (32000 x 256 -> 128)
// grid 250 x 256 threads (8 waves, 1 tile of 16 rows per wave). LDS 128KB dyn.
// ---------------------------------------------------------------------------
__global__ void encode_kernel(const float* __restrict__ x,
                              const float* __restrict__ w1,
                              const float* __restrict__ b1,
                              const float* __restrict__ w2,
                              const float* __restrict__ b2,
                              float* __restrict__ pe) {
  extern __shared__ char smem[];
  __bf16* w1s = (__bf16*)smem;              // swizzled [8kt][8nt][32][16]
  __bf16* w2s = w1s + FIN_ * DFF_;          // swizzled [4kt][8nt][32][16]
  __bf16* hid = w2s + DFF_ * DFF_;          // 8 waves x [16][128] row-major

  stage_b_swz(w1, w1s, FIN_, 8, DFF_);
  stage_b_swz(w2, w2s, DFF_, 8, DDIST_);
  __syncthreads();

  int wave = threadIdx.x >> 5, lane = threadIdx.x & 31;
  int hi = lane >> 4, li = lane & 15;
  int row0 = (blockIdx.x * 8 + wave) * 16;
  __bf16* myhid = hid + wave * 16 * DFF_;

  v8f acc[8] = {};
#pragma unroll
  for (int kt = 0; kt < FIN_ / 32; ++kt) {
    v16bf a = a_frag_f32(x + (size_t)row0 * FIN_, FIN_, kt * 32, lane);
#pragma unroll
    for (int nt = 0; nt < 8; ++nt)
      acc[nt] = wmma_bf16(a, b_frag_swz(w1s, 8, kt, nt, lane), acc[nt]);
  }
  // bias + elu -> LDS hidden (bf16)
#pragma unroll
  for (int nt = 0; nt < 8; ++nt) {
    int col = nt * 16 + li;
#pragma unroll
    for (int j = 0; j < 8; ++j) {
      int r = j + 8 * hi;
      myhid[r * DFF_ + col] = (__bf16)eluf(acc[nt][j] + b1[col]);
    }
  }
  __syncthreads();

  v8f acc2[8] = {};
#pragma unroll
  for (int kt = 0; kt < DFF_ / 32; ++kt) {
    v16bf a = a_frag_lds(myhid, DFF_, kt * 32, lane);
#pragma unroll
    for (int nt = 0; nt < 8; ++nt)
      acc2[nt] = wmma_bf16(a, b_frag_swz(w2s, 8, kt, nt, lane), acc2[nt]);
  }
#pragma unroll
  for (int nt = 0; nt < 8; ++nt) {
    int col = nt * 16 + li;
#pragma unroll
    for (int j = 0; j < 8; ++j) {
      int r = j + 8 * hi;
      pe[(size_t)(row0 + r) * DDIST_ + col] = acc2[nt][j] + b2[col];
    }
  }
}

// ---------------------------------------------------------------------------
// K2: LSH binning: bin = argmax([pe@rot[:, :8], -pe@rot[:, :8]])
// ---------------------------------------------------------------------------
__global__ void lsh_bin_kernel(const float* __restrict__ pe,
                               const float* __restrict__ rot,
                               int* __restrict__ bin_idx) {
  int g = blockIdx.x * blockDim.x + threadIdx.x;
  if (g >= B_ * N_) return;
  const float* p = pe + (size_t)g * DDIST_;
  float m[8] = {0, 0, 0, 0, 0, 0, 0, 0};
  for (int d = 0; d < DDIST_; ++d) {
    float pv = p[d];
    const float* rr = rot + d * ROTC_;
#pragma unroll
    for (int j = 0; j < 8; ++j) m[j] += pv * rr[j];
  }
  float best = m[0]; int bi = 0;
#pragma unroll
  for (int j = 1; j < 8; ++j) if (m[j] > best) { best = m[j]; bi = j; }
#pragma unroll
  for (int j = 0; j < 8; ++j) if (-m[j] > best) { best = -m[j]; bi = 8 + j; }
  bin_idx[g] = bi;
}

// ---------------------------------------------------------------------------
// K3: stable counting sort per batch (== jnp.argsort of bin_idx).
// Deterministic: thread t owns bin t and scans in index order.
// ---------------------------------------------------------------------------
__global__ void bin_sort_kernel(const int* __restrict__ bin_idx,
                                int* __restrict__ sorted,
                                int* __restrict__ bins_out) {
  int b = blockIdx.x, t = threadIdx.x;
  __shared__ int counts[NBINS_], starts[NBINS_];
  const int* bb = bin_idx + b * N_;
  if (t < NBINS_) {
    int c = 0;
    for (int i = 0; i < N_; ++i) c += (bb[i] == t);
    counts[t] = c;
  }
  __syncthreads();
  if (t == 0) {
    int s = 0;
    for (int k = 0; k < NBINS_; ++k) { starts[k] = s; s += counts[k]; }
  }
  __syncthreads();
  if (t < NBINS_) {
    int p = starts[t];
    for (int i = 0; i < N_; ++i)
      if (bb[i] == t) {
        sorted[b * N_ + p] = i;
        bins_out[b * N_ + p] = i;   // bins_split output (int32)
        ++p;
      }
  }
}

// ---------------------------------------------------------------------------
// K4: per-bin Gram matrix (WMMA, bf16) -> gaussian kernel -> row top-8.
// grid 64 (=B*NBINS), 256 threads. dyn LDS: parts 128KB + na 2KB + tiles 8KB.
// ---------------------------------------------------------------------------
__global__ void dist_topk_kernel(const float* __restrict__ pe,
                                 const int* __restrict__ sorted,
                                 int* __restrict__ esrc,
                                 int* __restrict__ edst,
                                 float* __restrict__ eval) {
  extern __shared__ char smem[];
  __bf16* parts = (__bf16*)smem;                       // [512][128]
  float*  na    = (float*)(smem + 512 * DDIST_ * 2);   // [512]
  float*  tiles = na + 512;                            // 8 x [16][16]

  int b = blockIdx.x >> 4, br = blockIdx.x & 15;
  const int* sb = sorted + b * N_ + br * BSZ_;

  for (int r = threadIdx.x; r < 512; r += blockDim.x) {
    float nn = 0.0f;
    v8bf* prow16 = (v8bf*)(parts + r * DDIST_);
    if (r < BSZ_) {
      const v4f* prow = (const v4f*)(pe + (size_t)(b * N_ + sb[r]) * DDIST_);
#pragma unroll 4
      for (int d8 = 0; d8 < DDIST_ / 8; ++d8) {
        v4f u = prow[2 * d8], v = prow[2 * d8 + 1];
        v8bf pk;
#pragma unroll
        for (int i = 0; i < 4; ++i) {
          nn += u[i] * u[i] + v[i] * v[i];
          pk[i] = (__bf16)u[i];
          pk[4 + i] = (__bf16)v[i];
        }
        prow16[d8] = pk;
      }
    } else {
      v8bf z = {};
#pragma unroll
      for (int d8 = 0; d8 < DDIST_ / 8; ++d8) prow16[d8] = z;
    }
    na[r] = nn;
  }
  __syncthreads();

  int wave = threadIdx.x >> 5, lane = threadIdx.x & 31;
  int hi = lane >> 4, li = lane & 15;
  float* mytile = tiles + wave * 256;

  for (int rr = 0; rr < 4; ++rr) {          // 32 row tiles / 8 waves
    int r0 = (wave + rr * 8) * 16;
    float tv[8]; int ti[8];
#pragma unroll
    for (int k = 0; k < 8; ++k) { tv[k] = -1e30f; ti[k] = 0; }

    for (int ct = 0; ct < 32; ++ct) {
      v8f acc = {};
#pragma unroll
      for (int kt = 0; kt < DDIST_ / 32; ++kt) {
        v16bf a  = a_frag_lds(parts + r0 * DDIST_, DDIST_, kt * 32, lane);
        v16bf bt = bT_frag_lds(parts, DDIST_, kt * 32, ct * 16, lane);
        acc = wmma_bf16(a, bt, acc);
      }
#pragma unroll
      for (int j = 0; j < 8; ++j) {
        int r = r0 + j + 8 * hi, c = ct * 16 + li;
        float d2 = na[r] - 2.0f * acc[j] + na[c];
        float dm = expf(-0.1f * sqrtf(fmaxf(d2, 1e-6f)));
        mytile[(j + 8 * hi) * 16 + li] = dm;
      }
      __syncthreads();
      if (lane < 16 && (r0 + lane) < BSZ_) {
#pragma unroll
        for (int cc = 0; cc < 16; ++cc) {
          int c = ct * 16 + cc;
          if (c < BSZ_) top8_insert(tv, ti, mytile[lane * 16 + cc], c);
        }
      }
      __syncthreads();
    }

    if (lane < 16 && (r0 + lane) < BSZ_) {
      int slot = br * BSZ_ + r0 + lane;
      int sg = sb[r0 + lane];
      size_t base = (size_t)(b * N_ + slot) * K_;
#pragma unroll
      for (int k = 0; k < K_; ++k) {
        esrc[base + k] = sg;
        edst[base + k] = sb[ti[k]];
        eval[base + k] = tv[k];
      }
    }
  }
}

// ---------------------------------------------------------------------------
// K5: analytic lexsort. Each src occurs exactly 8x (one row slot), so final
// position = b*64000 + src*8 + rank(dst). Sort 8 neighbors by dst.
// ---------------------------------------------------------------------------
__global__ void reorder_kernel(const int* __restrict__ esrc,
                               const int* __restrict__ edst,
                               const float* __restrict__ eval,
                               int* __restrict__ idx_out,
                               int* __restrict__ src2,
                               int* __restrict__ dst2,
                               float* __restrict__ val2) {
  int g = blockIdx.x * blockDim.x + threadIdx.x;
  if (g >= B_ * N_) return;
  int b = g / N_;
  size_t base = (size_t)g * K_;
  int s = esrc[base];
  int dd[8]; float vv[8];
#pragma unroll
  for (int k = 0; k < 8; ++k) { dd[k] = edst[base + k]; vv[k] = eval[base + k]; }
#pragma unroll
  for (int i = 0; i < 8; ++i)
#pragma unroll
    for (int j = 0; j < 7 - i; ++j)
      if (dd[j] > dd[j + 1]) {
        int t = dd[j]; dd[j] = dd[j + 1]; dd[j + 1] = t;
        float f = vv[j]; vv[j] = vv[j + 1]; vv[j + 1] = f;
      }
  size_t pos0 = (size_t)b * (N_ * K_) + (size_t)s * K_;
#pragma unroll
  for (int j = 0; j < 8; ++j) {
    size_t pos = pos0 + j;
    idx_out[3 * pos + 0] = b;
    idx_out[3 * pos + 1] = s;
    idx_out[3 * pos + 2] = dd[j];
    src2[pos] = s; dst2[pos] = dd[j]; val2[pos] = vv[j];
  }
}

// ---------------------------------------------------------------------------
// K6: edge FFN: sigmoid(elu([x_src|x_dst|val] @ W1 + b1) @ W2 + b2).
// 16-edge tile per wave; K=512 via WMMA (swizzled W1), col 512 via VALU FMA.
// grid 2000 x 256 threads. dyn LDS: W1 swz 128KB + row512 256B + hidden 64KB.
// ---------------------------------------------------------------------------
__global__ void edge_kernel(const float* __restrict__ x,
                            const float* __restrict__ w1,
                            const float* __restrict__ b1,
                            const float* __restrict__ w2,
                            const float* __restrict__ b2,
                            const int* __restrict__ src2,
                            const int* __restrict__ dst2,
                            const float* __restrict__ val2,
                            float* __restrict__ ev_out) {
  extern __shared__ char smem[];
  __bf16* w1s  = (__bf16*)smem;                         // swz [16kt][8nt][32][16]
  __bf16* wrow = w1s + 512 * DFF_;                      // w1 row 512 [128]
  float*  hid  = (float*)(smem + (512 * DFF_ + DFF_) * 2);  // 8 x [16][128]

  stage_b_swz(w1, w1s, 512, 8, DFF_);                   // rows 0..511
  for (int i = threadIdx.x; i < DFF_; i += blockDim.x)
    wrow[i] = (__bf16)w1[(size_t)512 * DFF_ + i];
  __syncthreads();

  int wave = threadIdx.x >> 5, lane = threadIdx.x & 31;
  int hi = lane >> 4, li = lane & 15;
  int e0 = (blockIdx.x * 8 + wave) * 16;
  float* myhid = hid + wave * 16 * DFF_;

  // this lane's edge row (A-fragment row = lane&15)
  int erow = e0 + li;
  int bb = erow / (N_ * K_);
  const float* xs = x + (size_t)(bb * N_ + src2[erow]) * FIN_;
  const float* xd = x + (size_t)(bb * N_ + dst2[erow]) * FIN_;
  __builtin_prefetch(xs, 0, 0);
  __builtin_prefetch(xd, 0, 0);

  v8f acc[8] = {};
#pragma unroll 4
  for (int kt = 0; kt < 16; ++kt) {           // K = 512 over [x_src | x_dst]
    int k0 = kt * 32;
    int kc0 = k0 + hi * 8, kc1 = kc0 + 16;
    const v4f* p0 = (const v4f*)((kc0 < FIN_) ? (xs + kc0) : (xd + kc0 - FIN_));
    const v4f* p1 = (const v4f*)((kc1 < FIN_) ? (xs + kc1) : (xd + kc1 - FIN_));
    v4f x0 = p0[0], x1 = p0[1], x2 = p1[0], x3 = p1[1];
    v16bf a;
#pragma unroll
    for (int i = 0; i < 4; ++i) {
      a[i]      = (__bf16)x0[i];
      a[4 + i]  = (__bf16)x1[i];
      a[8 + i]  = (__bf16)x2[i];
      a[12 + i] = (__bf16)x3[i];
    }
#pragma unroll
    for (int nt = 0; nt < 8; ++nt)
      acc[nt] = wmma_bf16(a, b_frag_swz(w1s, 8, kt, nt, lane), acc[nt]);
  }

  // fold in the 513th input column (kernel value) with VALU FMAs
  float vr[8];
#pragma unroll
  for (int j = 0; j < 8; ++j) vr[j] = val2[e0 + j + 8 * hi];
#pragma unroll
  for (int nt = 0; nt < 8; ++nt) {
    int col = nt * 16 + li;
    float wv = (float)wrow[col];
#pragma unroll
    for (int j = 0; j < 8; ++j) {
      int r = j + 8 * hi;
      myhid[r * DFF_ + col] = eluf(acc[nt][j] + vr[j] * wv + b1[col]);
    }
  }
  __syncthreads();

  if (lane < 16) {
    float s = 0.0f;
    for (int d = 0; d < DFF_; ++d) s += myhid[lane * DFF_ + d] * w2[d];
    float z = s + b2[0];
    ev_out[e0 + lane] = 1.0f / (1.0f + expf(-z));
  }
}

// ---------------------------------------------------------------------------
extern "C" void kernel_launch(void* const* d_in, const int* in_sizes, int n_in,
                              void* d_out, int out_size, void* d_ws,
                              size_t ws_size, hipStream_t stream) {
  (void)in_sizes; (void)n_in; (void)out_size; (void)ws_size;
  const float* inputs = (const float*)d_in[0];
  const float* enc_w1 = (const float*)d_in[1];
  const float* enc_b1 = (const float*)d_in[2];
  const float* enc_w2 = (const float*)d_in[3];
  const float* enc_b2 = (const float*)d_in[4];
  const float* edge_w1 = (const float*)d_in[5];
  const float* edge_b1 = (const float*)d_in[6];
  const float* edge_w2 = (const float*)d_in[7];
  const float* edge_b2 = (const float*)d_in[8];
  const float* rot = (const float*)d_in[9];

  // workspace carve (~20 MB)
  char* w = (char*)d_ws;
  float* pe = (float*)w;      w += (size_t)B_ * N_ * DDIST_ * 4;
  int* bin_idx = (int*)w;     w += (size_t)B_ * N_ * 4;
  int* sorted = (int*)w;      w += (size_t)B_ * N_ * 4;
  int* esrc = (int*)w;        w += (size_t)EDGES_ * 4;
  int* edst = (int*)w;        w += (size_t)EDGES_ * 4;
  float* eval = (float*)w;    w += (size_t)EDGES_ * 4;
  int* src2 = (int*)w;        w += (size_t)EDGES_ * 4;
  int* dst2 = (int*)w;        w += (size_t)EDGES_ * 4;
  float* val2 = (float*)w;    w += (size_t)EDGES_ * 4;

  // output carve: indices [E,3] i32 | edge_vals [E] f32 | bins_split i32
  int* idx_out = (int*)d_out;
  float* ev_out = (float*)d_out + (size_t)EDGES_ * 3;
  int* bins_out = (int*)d_out + (size_t)EDGES_ * 3 + EDGES_;

  size_t lds1 = (size_t)(FIN_ * DFF_ + DFF_ * DFF_ + 8 * 16 * DFF_) * 2; // 128KB
  size_t lds4 = (size_t)512 * DDIST_ * 2 + 512 * 4 + 8 * 256 * 4;        // ~138KB
  size_t lds6 = (size_t)(512 * DFF_ + DFF_) * 2 + 8 * 16 * DFF_ * 4;     // ~192KB

  encode_kernel<<<250, 256, lds1, stream>>>(inputs, enc_w1, enc_b1, enc_w2,
                                            enc_b2, pe);
  lsh_bin_kernel<<<(B_ * N_ + 255) / 256, 256, 0, stream>>>(pe, rot, bin_idx);
  bin_sort_kernel<<<B_, 32, 0, stream>>>(bin_idx, sorted, bins_out);
  dist_topk_kernel<<<B_ * NBINS_, 256, lds4, stream>>>(pe, sorted, esrc, edst,
                                                       eval);
  reorder_kernel<<<(B_ * N_ + 255) / 256, 256, 0, stream>>>(
      esrc, edst, eval, idx_out, src2, dst2, val2);
  edge_kernel<<<2000, 256, lds6, stream>>>(inputs, edge_w1, edge_b1, edge_w2,
                                           edge_b2, src2, dst2, val2, ev_out);
}
